// MultiHeadAttention_14766097564131
// MI455X (gfx1250) — matrix operand order repaired and checked by reading, not compiled, not run on hardware
//
#include <hip/hip_runtime.h>

// MI455X / gfx1250, wave32. bf16 WMMA + TDM async tile staging + NT mask stream.

typedef __attribute__((ext_vector_type(16))) __bf16 v16bf;
typedef __attribute__((ext_vector_type(8)))  float  v8f;
typedef __attribute__((ext_vector_type(4)))  float  v4f;

#define DEV static __device__ __forceinline__

constexpr int BB = 4, SS = 2048, DD = 1024, HHD = 16;   // batch, seq, model, heads

DEV unsigned short f2bfbits(float x) {
  unsigned u = __float_as_uint(x);
  u += 0x7FFFu + ((u >> 16) & 1u);          // round-to-nearest-even
  return (unsigned short)(u >> 16);
}
DEV __bf16 f2bf(float x) {
  unsigned short s = f2bfbits(x);
  return __builtin_bit_cast(__bf16, s);
}

union FragU { v16bf v; uint4 q[2]; };
DEV v16bf ldfrag(const __bf16* p0, const __bf16* p1) {
  FragU f;
  f.q[0] = *(const uint4*)p0;
  f.q[1] = *(const uint4*)p1;
  return f.v;
}

DEV v8f wmma_bf16(v16bf a, v16bf b, v8f c) {
  return __builtin_amdgcn_wmma_f32_16x16x32_bf16(false, a, false, b, (short)0, c, false, false);
}

// load 16 strided bf16 and pack into 8 dwords (for LDS-transposed staging)
DEV void ld16str(const __bf16* base, size_t stride, unsigned sh[8]) {
  #pragma unroll
  for (int j = 0; j < 16; ++j) {
    __bf16 v = base[(size_t)j * stride];
    unsigned short r = __builtin_bit_cast(unsigned short, v);
    if (j & 1) sh[j >> 1] |= ((unsigned)r) << 16; else sh[j >> 1] = (unsigned)r;
  }
}
DEV void st8lds(__bf16* dst, const unsigned sh[8]) {
  uint4 lo; lo.x = sh[0]; lo.y = sh[1]; lo.z = sh[2]; lo.w = sh[3];
  uint4 hi; hi.x = sh[4]; hi.y = sh[5]; hi.z = sh[6]; hi.w = sh[7];
  ((uint4*)dst)[0] = lo; ((uint4*)dst)[1] = hi;
}

DEV void wait_tensor0() {
#if __has_builtin(__builtin_amdgcn_s_wait_tensorcnt)
  __builtin_amdgcn_s_wait_tensorcnt((short)0);
#else
  asm volatile("s_wait_tensorcnt 0x0" ::: "memory");
#endif
}

// ---- Tensor Data Mover: 2D tile (rows x w_elems bf16, row stride in elems) -> LDS
#if __has_builtin(__builtin_amdgcn_tensor_load_to_lds)
#define HAVE_TDM 1
typedef __attribute__((ext_vector_type(4))) unsigned v4u;
typedef __attribute__((ext_vector_type(4))) int v4i;
typedef __attribute__((ext_vector_type(8))) int v8i;

DEV void tdm_load_2d(const void* lds_dst, const void* gsrc,
                     unsigned w_elems, unsigned rows, unsigned stride_elems) {
  unsigned long long ga = (unsigned long long)(size_t)gsrc;
  v4u g0;
  g0.x = 1u;                                            // count=1, user mode
  g0.y = (unsigned)(size_t)lds_dst;                     // LDS byte offset (addr[31:0])
  g0.z = (unsigned)ga;                                  // global addr [31:0]
  g0.w = (unsigned)((ga >> 32) & 0x01FFFFFFu) | (2u << 30);  // addr[56:32] | type=2
  v8i g1;
  g1[0] = (int)(1u << 16);                              // data_size = 2 bytes
  g1[1] = (int)(w_elems << 16);                         // tensor_dim0 [15:0]
  g1[2] = (int)(rows << 16);                            // dim0 hi=0 | tensor_dim1 [15:0]
  g1[3] = (int)(w_elems << 16);                         // dim1 hi=0 | tile_dim0
  g1[4] = (int)rows;                                    // tile_dim1 | tile_dim2=0
  g1[5] = (int)stride_elems;                            // tensor_dim0_stride [31:0]
  g1[6] = 0;
  g1[7] = 0;                                            // tensor_dim1_stride = 0 (2D)
  v4i z4 = {};
  v8i z8 = {};
  // 6-arg form (clang-23 / therock HIP headers): groups 0..3 + extra + cpol
  __builtin_amdgcn_tensor_load_to_lds(g0, g1, z4, z4, z8, 0);
}
#else
#define HAVE_TDM 0
#endif

// ---------------------------------------------------------------- fp32 -> bf16
__global__ void cvt_f32_bf16(const float* __restrict__ in, __bf16* __restrict__ out, int n4) {
  int i = blockIdx.x * blockDim.x + threadIdx.x;
  if (i >= n4) return;
  v4f f = __builtin_nontemporal_load((const v4f*)in + i);   // inputs read exactly once
  uint2 r;
  r.x = (unsigned)f2bfbits(f.x) | ((unsigned)f2bfbits(f.y) << 16);
  r.y = (unsigned)f2bfbits(f.z) | ((unsigned)f2bfbits(f.w) << 16);
  ((uint2*)out)[i] = r;
}

// ---------------------------------------------------------------- GEMM + bias
// C[M,N] = A[M,K] * B[K,N] + bias. WG tile 128x128, 8 waves of 64x32, K step 32.
// Double-buffered LDS; A tile via TDM (wave 0), B tile transposed by all threads.
template<bool OUT_BF16>
__global__ __launch_bounds__(256) void gemm_bias_kernel(
    const __bf16* __restrict__ Ag, const __bf16* __restrict__ Bm,
    const float* __restrict__ bias, void* __restrict__ Cout,
    int M, int N, int K)
{
  __shared__ __bf16 sA[2][128 * 32];   // [m][k]
  __shared__ __bf16 sB[2][128 * 32];   // [n][k] (transposed during staging)

  const int tid  = threadIdx.x;
  const int lane = tid & 31;
  const int w    = tid >> 5;
  const int l16  = lane & 15;
  const int hl   = lane >> 4;
  const int wm   = w >> 2, wn = w & 3;
  const int m0   = blockIdx.y * 128;
  const int n0   = blockIdx.x * 128;

  v8f acc[4][2];
  #pragma unroll
  for (int a = 0; a < 4; ++a)
    #pragma unroll
    for (int c = 0; c < 2; ++c) acc[a][c] = {};

  const int bn  = tid & 127;         // B staging column
  const int bkg = (tid >> 7) * 16;   // B staging k group (0/16)

  unsigned shb[8];
#if !HAVE_TDM
  uint4 ra[2];
#endif

  // ---- prologue: stage tile kk=0 into buffer 0
#if HAVE_TDM
  if (w == 0) tdm_load_2d(&sA[0][0], Ag + (size_t)m0 * K, 32, 128, (unsigned)K);
#else
  #pragma unroll
  for (int l = 0; l < 2; ++l) {
    int off = (tid + l * 256) * 8;
    int row = off >> 5, k = off & 31;
    *(uint4*)&sA[0][off] = *(const uint4*)(Ag + (size_t)(m0 + row) * K + k);
  }
#endif
  ld16str(Bm + (size_t)bkg * N + n0 + bn, (size_t)N, shb);
  st8lds(&sB[0][bn * 32 + bkg], shb);
#if HAVE_TDM
  if (w == 0) wait_tensor0();
#endif
  __syncthreads();

  int p = 0;
  for (int kk = 0; kk < K; kk += 32) {
    const bool next = (kk + 32) < K;
    if (next) {
#if HAVE_TDM
      if (w == 0)
        tdm_load_2d(&sA[p ^ 1][0], Ag + (size_t)m0 * K + kk + 32, 32, 128, (unsigned)K);
#else
      #pragma unroll
      for (int l = 0; l < 2; ++l) {
        int off = (tid + l * 256) * 8;
        int row = off >> 5, k = off & 31;
        ra[l] = *(const uint4*)(Ag + (size_t)(m0 + row) * K + kk + 32 + k);
      }
#endif
      ld16str(Bm + (size_t)(kk + 32 + bkg) * N + n0 + bn, (size_t)N, shb);
    }

    // compute from buffer p
    v16bf bfr[2];
    #pragma unroll
    for (int ni = 0; ni < 2; ++ni) {
      const __bf16* pb = &sB[p][(wn * 32 + ni * 16 + l16) * 32 + hl * 16];
      bfr[ni] = ldfrag(pb, pb + 8);
    }
    #pragma unroll
    for (int mi = 0; mi < 4; ++mi) {
      const __bf16* pa = &sA[p][(wm * 64 + mi * 16 + l16) * 32 + hl * 16];
      v16bf af = ldfrag(pa, pa + 8);
      #pragma unroll
      for (int ni = 0; ni < 2; ++ni)
        acc[mi][ni] = wmma_bf16(af, bfr[ni], acc[mi][ni]);
    }

    if (next) {
#if !HAVE_TDM
      #pragma unroll
      for (int l = 0; l < 2; ++l) {
        int off = (tid + l * 256) * 8;
        *(uint4*)&sA[p ^ 1][off] = ra[l];
      }
#endif
      st8lds(&sB[p ^ 1][bn * 32 + bkg], shb);
#if HAVE_TDM
      if (w == 0) wait_tensor0();
#endif
    }
    __syncthreads();
    p ^= 1;
  }

  // ---- epilogue: bias + store
  #pragma unroll
  for (int ni = 0; ni < 2; ++ni) {
    int col = n0 + wn * 32 + ni * 16 + l16;
    float bv = bias[col];
    #pragma unroll
    for (int mi = 0; mi < 4; ++mi)
      #pragma unroll
      for (int i = 0; i < 8; ++i) {
        int row = m0 + wm * 64 + mi * 16 + i + hl * 8;
        float v = acc[mi][ni][i] + bv;
        if (OUT_BF16) ((__bf16*)Cout)[(size_t)row * N + col] = f2bf(v);
        else __builtin_nontemporal_store(v, &((float*)Cout)[(size_t)row * N + col]);
      }
  }
}

// ---------------------------------------------------------------- flash attention
// Grid: (S/128, B*H). WG = 8 waves; wave owns 16 query rows, dh=64.
// K tile via TDM (double-buffered), V tile transposed by all threads, NT mask stream.
__global__ __launch_bounds__(256) void flash_attn_kernel(
    const __bf16* __restrict__ Q, const __bf16* __restrict__ Kb,
    const __bf16* __restrict__ Vb, const int* __restrict__ mask,
    __bf16* __restrict__ Ob)
{
  __shared__ __bf16 sK[2][64 * 64];     // [kv][dh]
  __shared__ __bf16 sV[2][64 * 64];     // [dh][kv] (transposed)
  __shared__ __bf16 sP[8][16 * 64];     // per-wave P tile [qrow][kv]

  const int tid = threadIdx.x, lane = tid & 31, w = tid >> 5;
  const int l16 = lane & 15, hl = lane >> 4;
  const int q0  = blockIdx.x * 128;
  const int b   = blockIdx.y >> 4;
  const int h   = blockIdx.y & 15;

  // Q fragments held in registers across the whole Sk loop
  const int qr = q0 + w * 16 + l16;
  const __bf16* qbase = Q + (size_t)(b * SS + qr) * DD + h * 64;
  v16bf qf[2];
  #pragma unroll
  for (int c = 0; c < 2; ++c) {
    const __bf16* pq = qbase + c * 32 + hl * 16;
    qf[c] = ldfrag(pq, pq + 8);
  }

  float mrow[8], lrow[8];
  #pragma unroll
  for (int i = 0; i < 8; ++i) { mrow[i] = -1e30f; lrow[i] = 0.f; }
  v8f acc[4];
  #pragma unroll
  for (int d = 0; d < 4; ++d) acc[d] = {};

  const size_t HS = (size_t)HHD * SS;
  const size_t mbase = ((size_t)(b * SS + q0 + w * 16)) * HS + (size_t)h * SS + l16;

  const int vdc = tid & 63;
  const int vkg = (tid >> 6) * 16;
  const __bf16* Kbase = Kb + (size_t)(b * SS) * DD + h * 64;
  const __bf16* Vbase = Vb + (size_t)(b * SS) * DD + h * 64 + vdc;

  unsigned shv[8];

  // ---- prologue: stage kv tile 0 into buffer 0
#if HAVE_TDM
  if (w == 0) tdm_load_2d(&sK[0][0], Kbase, 64, 64, DD);
#else
  #pragma unroll
  for (int l = 0; l < 2; ++l) {
    int off = (tid + l * 256) * 8;
    int kv = off >> 6, dc = off & 63;
    *(uint4*)&sK[0][off] = *(const uint4*)(Kbase + (size_t)kv * DD + dc);
  }
#endif
  ld16str(Vbase + (size_t)vkg * DD, (size_t)DD, shv);
  st8lds(&sV[0][vdc * 64 + vkg], shv);
#if HAVE_TDM
  if (w == 0) wait_tensor0();
#endif
  __syncthreads();

  int p = 0;
  for (int kv0 = 0; kv0 < SS; kv0 += 64) {
    const bool next = (kv0 + 64) < SS;
    if (next) {
#if HAVE_TDM
      if (w == 0) tdm_load_2d(&sK[p ^ 1][0], Kbase + (size_t)(kv0 + 64) * DD, 64, 64, DD);
#else
      #pragma unroll
      for (int l = 0; l < 2; ++l) {
        int off = (tid + l * 256) * 8;
        int kv = off >> 6, dc = off & 63;
        *(uint4*)&sK[p ^ 1][off] =
            *(const uint4*)(Kbase + (size_t)(kv0 + 64 + kv) * DD + dc);
      }
#endif
      ld16str(Vbase + (size_t)(kv0 + 64 + vkg) * DD, (size_t)DD, shv);
    }

    // scores: 4 tiles of 16x16, contraction dh=64 (2 WMMAs each)
    float st[4][8];
    #pragma unroll
    for (int j = 0; j < 4; ++j) {
      v8f s = {};
      #pragma unroll
      for (int c = 0; c < 2; ++c) {
        const __bf16* pk = &sK[p][(j * 16 + l16) * 64 + c * 32 + hl * 16];
        v16bf kf = ldfrag(pk, pk + 8);
        s = wmma_bf16(qf[c], kf, s);
      }
      #pragma unroll
      for (int i = 0; i < 8; ++i) {
        float sv = s[i] * 0.125f;   // 1/sqrt(64)
        int mv = __builtin_nontemporal_load(
            mask + mbase + (size_t)(i + hl * 8) * HS + kv0 + j * 16);
        st[j][i] = mv ? -1e9f : sv;
      }
    }
    // online softmax (row stats live in the 16-lane half-wave holding that row)
    #pragma unroll
    for (int i = 0; i < 8; ++i) {
      float mx = fmaxf(fmaxf(st[0][i], st[1][i]), fmaxf(st[2][i], st[3][i]));
      #pragma unroll
      for (int o = 1; o < 16; o <<= 1) mx = fmaxf(mx, __shfl_xor(mx, o, 32));
      float mnew  = fmaxf(mrow[i], mx);
      float alpha = __expf(mrow[i] - mnew);
      mrow[i] = mnew;
      float ss = 0.f;
      #pragma unroll
      for (int j = 0; j < 4; ++j) { float pe = __expf(st[j][i] - mnew); st[j][i] = pe; ss += pe; }
      #pragma unroll
      for (int o = 1; o < 16; o <<= 1) ss += __shfl_xor(ss, o, 32);
      lrow[i] = lrow[i] * alpha + ss;
      #pragma unroll
      for (int d = 0; d < 4; ++d) acc[d][i] *= alpha;
    }
    // restage P (C layout -> A layout) through per-wave LDS (DS in-order per wave)
    #pragma unroll
    for (int j = 0; j < 4; ++j)
      #pragma unroll
      for (int i = 0; i < 8; ++i)
        sP[w][(i + hl * 8) * 64 + j * 16 + l16] = f2bf(st[j][i]);
    // O += P x V  (contraction kv=64, N = dh)
    #pragma unroll
    for (int c = 0; c < 2; ++c) {
      const __bf16* pp = &sP[w][l16 * 64 + c * 32 + hl * 16];
      v16bf pf = ldfrag(pp, pp + 8);
      #pragma unroll
      for (int d = 0; d < 4; ++d) {
        const __bf16* pv = &sV[p][(d * 16 + l16) * 64 + c * 32 + hl * 16];
        v16bf vf = ldfrag(pv, pv + 8);
        acc[d] = wmma_bf16(pf, vf, acc[d]);
      }
    }

    if (next) {
      st8lds(&sV[p ^ 1][vdc * 64 + vkg], shv);
#if HAVE_TDM
      if (w == 0) wait_tensor0();
#endif
    }
    __syncthreads();
    p ^= 1;
  }

  #pragma unroll
  for (int d = 0; d < 4; ++d)
    #pragma unroll
    for (int i = 0; i < 8; ++i) {
      float o = acc[d][i] / lrow[i];
      Ob[(size_t)(b * SS + q0 + w * 16 + i + hl * 8) * DD + h * 64 + d * 16 + l16] = f2bf(o);
    }
}

// ---------------------------------------------------------------- launcher
extern "C" void kernel_launch(void* const* d_in, const int* in_sizes, int n_in,
                              void* d_out, int out_size, void* d_ws, size_t ws_size,
                              hipStream_t stream) {
  (void)in_sizes; (void)n_in; (void)out_size; (void)ws_size;

  const float* query = (const float*)d_in[0];
  const float* key   = (const float*)d_in[1];
  const float* value = (const float*)d_in[2];
  const int*   mask  = (const int*)d_in[3];
  const float* Wq = (const float*)d_in[4];  const float* bq = (const float*)d_in[5];
  const float* Wk = (const float*)d_in[6];  const float* bk = (const float*)d_in[7];
  const float* Wv = (const float*)d_in[8];  const float* bv = (const float*)d_in[9];
  const float* Wo = (const float*)d_in[10]; const float* bo = (const float*)d_in[11];

  constexpr size_t ACT = (size_t)BB * SS * DD;   // 8388608 elems
  constexpr size_t WEL = (size_t)DD * DD;        // 1048576 elems

  char* ws = (char*)d_ws;
  size_t off = 0;
  auto take = [&](size_t elems) { __bf16* p = (__bf16*)(ws + off); off += elems * 2; return p; };
  __bf16* xq = take(ACT); __bf16* xk = take(ACT); __bf16* xv = take(ACT);
  __bf16* wq = take(WEL); __bf16* wk = take(WEL); __bf16* wv = take(WEL); __bf16* wo = take(WEL);
  __bf16* Qp = take(ACT); __bf16* Kp = take(ACT); __bf16* Vp = take(ACT);
  __bf16* at = take(ACT);
  // total workspace use: ~120 MB

  cvt_f32_bf16<<<ACT / 4 / 256, 256, 0, stream>>>(query, xq, (int)(ACT / 4));
  cvt_f32_bf16<<<ACT / 4 / 256, 256, 0, stream>>>(key,   xk, (int)(ACT / 4));
  cvt_f32_bf16<<<ACT / 4 / 256, 256, 0, stream>>>(value, xv, (int)(ACT / 4));
  cvt_f32_bf16<<<WEL / 4 / 256, 256, 0, stream>>>(Wq, wq, (int)(WEL / 4));
  cvt_f32_bf16<<<WEL / 4 / 256, 256, 0, stream>>>(Wk, wk, (int)(WEL / 4));
  cvt_f32_bf16<<<WEL / 4 / 256, 256, 0, stream>>>(Wv, wv, (int)(WEL / 4));
  cvt_f32_bf16<<<WEL / 4 / 256, 256, 0, stream>>>(Wo, wo, (int)(WEL / 4));

  const int M = BB * SS;                         // 8192
  dim3 gg(DD / 128, M / 128);                    // (8, 64)
  gemm_bias_kernel<true ><<<gg, 256, 0, stream>>>(xq, wq, bq, Qp, M, DD, DD);
  gemm_bias_kernel<true ><<<gg, 256, 0, stream>>>(xk, wk, bk, Kp, M, DD, DD);
  gemm_bias_kernel<true ><<<gg, 256, 0, stream>>>(xv, wv, bv, Vp, M, DD, DD);

  flash_attn_kernel<<<dim3(SS / 128, BB * HHD), 256, 0, stream>>>(Qp, Kp, Vp, mask, at);

  gemm_bias_kernel<false><<<gg, 256, 0, stream>>>(at, wo, bo, d_out, M, DD, DD);
}
